// SelfAttention_88132728914454
// MI455X (gfx1250) — matrix operand order repaired and checked by reading, not compiled, not run on hardware
//
#include <hip/hip_runtime.h>
#include <cstdint>
#include <cstddef>

typedef __bf16 bf16;
typedef __attribute__((ext_vector_type(16))) __bf16 v16bf;
typedef __attribute__((ext_vector_type(8)))  __bf16 v8bf;
typedef __attribute__((ext_vector_type(8)))  float  v8f;

#define WMMA_BF16(a, b, c) \
  __builtin_amdgcn_wmma_f32_16x16x32_bf16(false, (a), false, (b), (short)0, (c), false, false)

// --- CDNA5 async global->LDS path (gate on toolchain support) --------------
#ifdef __has_builtin
#if __has_builtin(__builtin_amdgcn_global_load_async_to_lds_b128)
#define HAVE_ASYNC_LDS 1
#endif
#if __has_builtin(__builtin_amdgcn_s_wait_asynccnt)
#define HAVE_WAIT_ASYNC 1
#endif
#endif
#ifndef HAVE_ASYNC_LDS
#define HAVE_ASYNC_LDS 0
#endif
#ifndef HAVE_WAIT_ASYNC
#define HAVE_WAIT_ASYNC 0
#endif

// Builtin prototype (from hipcc diagnostic): v4i AS1*, v4i AS3*, Ii, Ii
typedef int i32x4 __attribute__((vector_size(16)));
typedef __attribute__((address_space(1))) i32x4 as1_i32x4;
typedef __attribute__((address_space(3))) i32x4 as3_i32x4;

template <int N>
__device__ __forceinline__ void wait_async() {
#if HAVE_WAIT_ASYNC
  __builtin_amdgcn_s_wait_asynccnt(N);
#else
  if constexpr (N == 0)      asm volatile("s_wait_asynccnt 0" ::: "memory");
  else if constexpr (N == 4) asm volatile("s_wait_asynccnt 4" ::: "memory");
  else                       asm volatile("s_wait_asynccnt 8" ::: "memory");
#endif
}

// 16-byte global->LDS copy: async DMA when available, sync fallback otherwise.
__device__ __forceinline__ void copy16_g2l(const bf16* g, bf16* l) {
#if HAVE_ASYNC_LDS
  __builtin_amdgcn_global_load_async_to_lds_b128(
      (as1_i32x4*)g, (as3_i32x4*)l, 0, 0);
#else
  *(uint4*)l = *(const uint4*)g;  // covered by dscnt fence at __syncthreads
#endif
}

// Problem constants (match reference setup_inputs)
constexpr int Bsz = 2, S = 2048, D = 1024, H = 16, DH = 64;
constexpr int NT = Bsz * S;   // 4096 tokens
constexpr int N3 = 3 * D;     // 3072

// Workspace layout (bf16 element offsets), total ~42 MB
constexpr size_t OWQ = 0;                              // wqkvT  [3D][D]
constexpr size_t OWO = OWQ + (size_t)N3 * D;           // woutT  [D][D]
constexpr size_t OQ  = OWO + (size_t)D * D;            // Q      [B*H][S][DH]
constexpr size_t OK_ = OQ  + (size_t)Bsz * H * S * DH; // K      [B*H][S][DH]
constexpr size_t OV  = OK_ + (size_t)Bsz * H * S * DH; // V^T    [B*H][DH][S]
constexpr size_t OA  = OV  + (size_t)Bsz * H * S * DH; // attn   [NT][D]

__device__ __forceinline__ bf16 f2bf(float f) {
  union { float f; uint32_t u; } in; in.f = f;
  uint32_t u = in.u;
  uint32_t r = (u + 0x7fffu + ((u >> 16) & 1u)) >> 16;  // RNE
  union { uint16_t s; bf16 b; } out; out.s = (uint16_t)r;
  return out.b;
}

__device__ __forceinline__ v8f vzero() {
  v8f r;
#pragma unroll
  for (int i = 0; i < 8; ++i) r[i] = 0.f;
  return r;
}

// Build a 16x32 A-frag / 32x16 B-frag from two contiguous 8-element chunks.
__device__ __forceinline__ v16bf make_frag(const bf16* p0, const bf16* p1) {
  v8bf lo = *(const v8bf*)p0;
  v8bf hi = *(const v8bf*)p1;
  v16bf r;
#pragma unroll
  for (int i = 0; i < 8; ++i) { r[i] = lo[i]; r[i + 8] = hi[i]; }
  return r;
}

// ---------------- weight convert + transpose (fp32 [r][c] -> bf16 [c][r]) --
__global__ void cvt_transpose_kernel(const float* __restrict__ src,
                                     bf16* __restrict__ dst, int rows, int cols) {
  int i = blockIdx.x * blockDim.x + threadIdx.x;
  if (i < rows * cols) {
    int r = i / cols, c = i % cols;
    dst[(size_t)c * rows + r] = f2bf(src[i]);
  }
}

// ---------------- double-buffered 64x64-tile GEMM core ---------------------
// C = A(MxK) * Bt^T, K-step 64, async LDS fill overlapped with WMMA compute.
constexpr int KSTEP = 64;
constexpr int LDSS  = 80;                 // 64 K-elems + 16 pad (160B rows)
constexpr int TILE_E = 64 * LDSS;         // LDS elems per matrix per buffer

template <bool AF32>
__device__ __forceinline__ void gemm_core_64x64(
    const void* __restrict__ Ap, const bf16* __restrict__ Bt, int Kdim,
    int mBase, int nBase, bf16* AsBase, bf16* BsBase, v8f (&acc)[2][2]) {
  const int tid  = threadIdx.x;
  const int lane = tid & 31;
  const int wave = tid >> 5;
  const int wrow = (wave >> 1) * 32;
  const int wcol = (wave & 1) * 32;
  const int n    = lane & 15;
  const int g    = lane >> 4;

  // Stage one 64(M/N) x 64(K) bf16 tile pair into LDS buffer `buf`.
  auto issue = [&](int k0, int buf) {
    bf16* As = AsBase + buf * TILE_E;
    bf16* Bs = BsBase + buf * TILE_E;
#pragma unroll
    for (int pass = 0; pass < 4; ++pass) {
      const int chunk = tid + pass * 128;   // 512 16B-chunks per tile
      const int row = chunk >> 3;
      const int col = (chunk & 7) * 8;
      if constexpr (AF32) {                 // fp32 source: convert in VGPRs
        const float* A = (const float*)Ap;
        const float4 f0 = *(const float4*)&A[(size_t)(mBase + row) * Kdim + k0 + col];
        const float4 f1 = *(const float4*)&A[(size_t)(mBase + row) * Kdim + k0 + col + 4];
        bf16* d = &As[row * LDSS + col];
        d[0] = f2bf(f0.x); d[1] = f2bf(f0.y); d[2] = f2bf(f0.z); d[3] = f2bf(f0.w);
        d[4] = f2bf(f1.x); d[5] = f2bf(f1.y); d[6] = f2bf(f1.z); d[7] = f2bf(f1.w);
      } else {                              // bf16 source: async DMA to LDS
        copy16_g2l((const bf16*)Ap + (size_t)(mBase + row) * Kdim + k0 + col,
                   &As[row * LDSS + col]);
      }
      copy16_g2l(&Bt[(size_t)(nBase + row) * Kdim + k0 + col],
                 &Bs[row * LDSS + col]);
    }
  };
  constexpr int NASY = AF32 ? 4 : 8;        // async instrs per issue per wave

  issue(0, 0);
  for (int k0 = 0; k0 < Kdim; k0 += KSTEP) {
    const int buf = (k0 / KSTEP) & 1;
    const bool more = (k0 + KSTEP) < Kdim;
    if (more) issue(k0 + KSTEP, buf ^ 1);   // prefetch next tile pair
    if (more) wait_async<NASY>();           // current tile's DMA complete
    else      wait_async<0>();
    __syncthreads();                        // fences dscnt for AF32 stores

    const bf16* As = AsBase + buf * TILE_E;
    const bf16* Bs = BsBase + buf * TILE_E;
#pragma unroll
    for (int kk = 0; kk < 2; ++kk) {        // two 16x16x32 sub-steps
      v16bf af[2], bfg[2];
#pragma unroll
      for (int rt = 0; rt < 2; ++rt) {
        const bf16* p = &As[(wrow + rt * 16 + n) * LDSS + kk * 32];
        af[rt] = make_frag(p + g * 8, p + 16 + g * 8);
      }
#pragma unroll
      for (int ct = 0; ct < 2; ++ct) {
        const bf16* p = &Bs[(wcol + ct * 16 + n) * LDSS + kk * 32];
        bfg[ct] = make_frag(p + g * 8, p + 16 + g * 8);
      }
#pragma unroll
      for (int rt = 0; rt < 2; ++rt)
#pragma unroll
        for (int ct = 0; ct < 2; ++ct)
          acc[rt][ct] = WMMA_BF16(af[rt], bfg[ct], acc[rt][ct]);
    }
    __syncthreads();                        // reads done before buf reuse
  }
}

// ---------------- kernel 2: QKV projection, scatter into per-head Q/K/V^T --
__global__ __launch_bounds__(128) void gemm_qkv_kernel(
    const float* __restrict__ X, const bf16* __restrict__ WqT,
    const float* __restrict__ bias,
    bf16* __restrict__ Qo, bf16* __restrict__ Ko, bf16* __restrict__ Vto) {
  __shared__ bf16 As[2 * TILE_E];
  __shared__ bf16 Bs[2 * TILE_E];
  const int mBase = blockIdx.y * 64;
  const int nBase = blockIdx.x * 64;
  v8f acc[2][2] = {{vzero(), vzero()}, {vzero(), vzero()}};
  gemm_core_64x64<true>(X, WqT, D, mBase, nBase, As, Bs, acc);

  const int lane = threadIdx.x & 31;
  const int wave = threadIdx.x >> 5;
  const int wrow = (wave >> 1) * 32, wcol = (wave & 1) * 32;
  const int n = lane & 15, g = lane >> 4;
#pragma unroll
  for (int rt = 0; rt < 2; ++rt)
#pragma unroll
    for (int ct = 0; ct < 2; ++ct) {
      const int m0 = mBase + wrow + rt * 16 + g * 8;
      const int c0 = nBase + wcol + ct * 16;  // 16-col tile lies in one 64-seg
      const int h = c0 / (3 * DH);
      const int r = c0 % (3 * DH);
      const int which = r / DH;
      const int d = (r % DH) + n;
      const float bv = bias[c0 + n];
#pragma unroll
      for (int i = 0; i < 8; ++i) {
        const int t = m0 + i;
        const int b = t >> 11;        // / S
        const int s = t & (S - 1);
        const bf16 o = f2bf(acc[rt][ct][i] + bv);
        const size_t bh = (size_t)(b * H + h);
        if (which == 0)      Qo [(bh * S + s) * DH + d] = o;
        else if (which == 1) Ko [(bh * S + s) * DH + d] = o;
        else                 Vto[(bh * DH + d) * S + s] = o;
      }
    }
}

// ---------------- kernel 3: flash attention, one wave = 16 query rows ------
__global__ __launch_bounds__(256) void attn_kernel(
    const bf16* __restrict__ Q, const bf16* __restrict__ K,
    const bf16* __restrict__ Vt, const float* __restrict__ mask,
    bf16* __restrict__ out) {
  __shared__ bf16 Plds[8][16 * 72];  // per-wave P staging, rows 16B aligned
  const int bh   = blockIdx.y;       // b*H + h
  const int tid  = threadIdx.x;
  const int wave = tid >> 5;
  const int lane = tid & 31;
  const int n = lane & 15, g = lane >> 4;
  const int qrow0 = blockIdx.x * 128 + wave * 16;

  const bf16* Qb = Q  + ((size_t)bh * S + qrow0) * DH;
  const bf16* Kb = K  + (size_t)bh * S * DH;
  const bf16* Vb = Vt + (size_t)bh * DH * S;

  // Q A-frags (row = qrow0 + n), loaded once, reused for all key blocks
  const bf16* qrp = Qb + (size_t)n * DH;
  const v16bf qa0 = make_frag(qrp + g * 8,      qrp + 16 + g * 8);
  const v16bf qa1 = make_frag(qrp + 32 + g * 8, qrp + 48 + g * 8);

  v8f acc[4] = {vzero(), vzero(), vzero(), vzero()};
  float mrow[8], lrow[8];
#pragma unroll
  for (int i = 0; i < 8; ++i) { mrow[i] = -1e30f; lrow[i] = 0.f; }

  bf16* Pl = &Plds[wave][0];
  constexpr float inv_scale = 0.125f;  // 1/sqrt(D/H) = 1/8

  for (int kb = 0; kb < S; kb += 64) {
    // prefetch next key-block's mask rows (rows 0..15 split by half-wave)
    if (kb + 64 < S)
      __builtin_prefetch(&mask[(size_t)(qrow0 + n) * S + kb + 64 + g * 32], 0, 0);
    // ---- scores: Q(16x64) . K^T -> 4 tiles of 16x16
    v8f st[4];
#pragma unroll
    for (int nt = 0; nt < 4; ++nt) {
      const bf16* krow = Kb + (size_t)(kb + nt * 16 + n) * DH;
      v16bf kf0 = make_frag(krow + g * 8,      krow + 16 + g * 8);
      v16bf kf1 = make_frag(krow + 32 + g * 8, krow + 48 + g * 8);
      v8f s = vzero();
      s = WMMA_BF16(qa0, kf0, s);
      s = WMMA_BF16(qa1, kf1, s);
      st[nt] = s;
    }
    // ---- scale + mask
#pragma unroll
    for (int nt = 0; nt < 4; ++nt)
#pragma unroll
      for (int i = 0; i < 8; ++i) {
        const int srow = qrow0 + g * 8 + i;
        const int key  = kb + nt * 16 + n;
        st[nt][i] = st[nt][i] * inv_scale + mask[(size_t)srow * S + key];
      }
    // ---- online softmax: row max via 16-lane butterflies
    float mnew[8], corr[8];
#pragma unroll
    for (int i = 0; i < 8; ++i) {
      float mx = fmaxf(fmaxf(st[0][i], st[1][i]), fmaxf(st[2][i], st[3][i]));
#pragma unroll
      for (int w = 1; w <= 8; w <<= 1) mx = fmaxf(mx, __shfl_xor(mx, w, 32));
      mnew[i] = fmaxf(mrow[i], mx);
      corr[i] = __expf(mrow[i] - mnew[i]);
      mrow[i] = mnew[i];
      lrow[i] *= corr[i];
    }
#pragma unroll
    for (int dt = 0; dt < 4; ++dt)
#pragma unroll
      for (int i = 0; i < 8; ++i) acc[dt][i] *= corr[i];

    float rsum[8];
#pragma unroll
    for (int i = 0; i < 8; ++i) rsum[i] = 0.f;
#pragma unroll
    for (int nt = 0; nt < 4; ++nt)
#pragma unroll
      for (int i = 0; i < 8; ++i) {
        float p = __expf(st[nt][i] - mnew[i]);
        st[nt][i] = p;
        rsum[i] += p;
      }
#pragma unroll
    for (int i = 0; i < 8; ++i) {
      float sm = rsum[i];
#pragma unroll
      for (int w = 1; w <= 8; w <<= 1) sm += __shfl_xor(sm, w, 32);
      lrow[i] += sm;
    }
    // ---- stage P: C-layout -> LDS row-major -> A-frags (same-wave LDS,
    //      ordered by DScnt; explicit CDNA5 wait for cross-lane visibility)
#pragma unroll
    for (int nt = 0; nt < 4; ++nt)
#pragma unroll
      for (int i = 0; i < 8; ++i)
        Pl[(g * 8 + i) * 72 + nt * 16 + n] = f2bf(st[nt][i]);
    asm volatile("s_wait_dscnt 0" ::: "memory");
    const bf16* prow = Pl + n * 72;
    v16bf pa0 = make_frag(prow + g * 8,      prow + 16 + g * 8);
    v16bf pa1 = make_frag(prow + 32 + g * 8, prow + 48 + g * 8);
    // ---- acc += P(16x64) @ V(64x64); V^T rows are contiguous along keys
#pragma unroll
    for (int dt = 0; dt < 4; ++dt) {
      const bf16* vrow = Vb + (size_t)(dt * 16 + n) * S + kb;
      v16bf vf0 = make_frag(vrow + g * 8,      vrow + 16 + g * 8);
      v16bf vf1 = make_frag(vrow + 32 + g * 8, vrow + 48 + g * 8);
      acc[dt] = WMMA_BF16(pa0, vf0, acc[dt]);
      acc[dt] = WMMA_BF16(pa1, vf1, acc[dt]);
    }
  }
  // ---- normalize and write token-major attn output (bf16)
  const int b = bh >> 4, h = bh & 15;
#pragma unroll
  for (int dt = 0; dt < 4; ++dt)
#pragma unroll
    for (int i = 0; i < 8; ++i) {
      const int srow = qrow0 + g * 8 + i;
      const float o = acc[dt][i] / lrow[i];
      out[((size_t)(b * S + srow)) * D + h * DH + dt * 16 + n] = f2bf(o);
    }
}

// ---------------- kernel 4: output projection (+bias), fp32 out -----------
__global__ __launch_bounds__(128) void gemm_out_kernel(
    const bf16* __restrict__ Aattn, const bf16* __restrict__ WoT,
    const float* __restrict__ bias, float* __restrict__ out) {
  __shared__ bf16 As[2 * TILE_E];
  __shared__ bf16 Bs[2 * TILE_E];
  const int mBase = blockIdx.y * 64;
  const int nBase = blockIdx.x * 64;
  v8f acc[2][2] = {{vzero(), vzero()}, {vzero(), vzero()}};
  gemm_core_64x64<false>(Aattn, WoT, D, mBase, nBase, As, Bs, acc);

  const int lane = threadIdx.x & 31;
  const int wave = threadIdx.x >> 5;
  const int wrow = (wave >> 1) * 32, wcol = (wave & 1) * 32;
  const int n = lane & 15, g = lane >> 4;
#pragma unroll
  for (int rt = 0; rt < 2; ++rt)
#pragma unroll
    for (int ct = 0; ct < 2; ++ct) {
      const int m0 = mBase + wrow + rt * 16 + g * 8;
      const int c  = nBase + wcol + ct * 16 + n;
      const float bv = bias[c];
#pragma unroll
      for (int i = 0; i < 8; ++i)
        out[(size_t)(m0 + i) * D + c] = acc[rt][ct][i] + bv;
    }
}

// ---------------- launch ---------------------------------------------------
extern "C" void kernel_launch(void* const* d_in, const int* in_sizes, int n_in,
                              void* d_out, int out_size, void* d_ws, size_t ws_size,
                              hipStream_t stream) {
  const float* x     = (const float*)d_in[0];
  const float* mask  = (const float*)d_in[1];
  const float* w_qkv = (const float*)d_in[2];
  const float* b_qkv = (const float*)d_in[3];
  const float* w_out = (const float*)d_in[4];
  const float* b_out = (const float*)d_in[5];
  float* out = (float*)d_out;

  bf16* ws   = (bf16*)d_ws;
  bf16* wqT  = ws + OWQ;
  bf16* woT  = ws + OWO;
  bf16* Qw   = ws + OQ;
  bf16* Kw   = ws + OK_;
  bf16* Vtw  = ws + OV;
  bf16* attn = ws + OA;

  // 1) weight convert + transpose (bf16, B^T layout for WMMA B-frags)
  cvt_transpose_kernel<<<(N3 * D + 255) / 256, 256, 0, stream>>>(w_qkv, wqT, D, N3);
  cvt_transpose_kernel<<<(D * D + 255) / 256, 256, 0, stream>>>(w_out, woT, D, D);
  // 2) QKV projection -> per-head Q, K, V^T (bf16)
  gemm_qkv_kernel<<<dim3(N3 / 64, NT / 64), 128, 0, stream>>>(x, wqT, b_qkv, Qw, Kw, Vtw);
  // 3) flash attention
  attn_kernel<<<dim3(S / 128, Bsz * H), 256, 0, stream>>>(Qw, Kw, Vtw, mask, attn);
  // 4) output projection -> fp32 d_out
  gemm_out_kernel<<<dim3(D / 64, NT / 64), 128, 0, stream>>>(attn, woT, b_out, out);
}
